// sigMoE_22093311771200
// MI455X (gfx1250) — matrix operand order repaired and checked by reading, not compiled, not run on hardware
//
#include <hip/hip_runtime.h>
#include <hip/hip_bf16.h>
#include <math.h>

#define HDIM 1024
#define FDIM 4096
#define NEXP 8
#define NTOK 4096          // B*S
#define NPAIR (2*NTOK)     // top-2 -> 2 pairs per token

typedef __attribute__((ext_vector_type(16))) __bf16 v16bf;
typedef __attribute__((ext_vector_type(8)))  float  v8f;

union FragU {
    uint4 u[2];   // 32 bytes = 16 bf16
    v16bf v;
};

__device__ __forceinline__ unsigned short f2bf(float f) {
    unsigned int u = __float_as_uint(f);
    u += 0x7fffu + ((u >> 16) & 1u);   // round-to-nearest-even
    return (unsigned short)(u >> 16);
}

// ---------------------------------------------------------------------------
// Router: one wave32 per token. scores = x . router_w + b; top-2; normalized
// pair probs; append (2*t+j) to per-expert list.
// ---------------------------------------------------------------------------
__global__ __launch_bounds__(256) void router_kernel(
    const float* __restrict__ x, const float* __restrict__ rw,
    const float* __restrict__ rb, int* __restrict__ cnt,
    int* __restrict__ list, float* __restrict__ prob)
{
    const int wid  = threadIdx.x >> 5;
    const int lane = threadIdx.x & 31;
    const int t    = blockIdx.x * 8 + wid;

    float acc[NEXP];
#pragma unroll
    for (int e = 0; e < NEXP; ++e) acc[e] = 0.f;

    const float* xr = x + (size_t)t * HDIM;
    for (int h = lane; h < HDIM; h += 32) {
        const float xv = xr[h];
        const float4* r4 = (const float4*)(rw + (size_t)h * NEXP);
        const float4 a = r4[0], b = r4[1];
        acc[0] += xv * a.x; acc[1] += xv * a.y; acc[2] += xv * a.z; acc[3] += xv * a.w;
        acc[4] += xv * b.x; acc[5] += xv * b.y; acc[6] += xv * b.z; acc[7] += xv * b.w;
    }
#pragma unroll
    for (int off = 16; off > 0; off >>= 1)
#pragma unroll
        for (int e = 0; e < NEXP; ++e) acc[e] += __shfl_xor(acc[e], off, 32);

    if (lane == 0) {
        float s[NEXP];
#pragma unroll
        for (int e = 0; e < NEXP; ++e) s[e] = acc[e] + rb[e];
        int e0 = 0;
#pragma unroll
        for (int e = 1; e < NEXP; ++e) if (s[e] > s[e0]) e0 = e;   // first-index tie break
        int e1 = (e0 == 0) ? 1 : 0;
#pragma unroll
        for (int e = 0; e < NEXP; ++e) if (e != e0 && s[e] > s[e1]) e1 = e;
        // normalized top-2 probs: softmax-Z cancels
        const float b  = expf(s[e1] - s[e0]);      // <= 1
        const float p0 = 1.f / (1.f + b);
        const float p1 = b   / (1.f + b);
        const int pos0 = atomicAdd(&cnt[e0], 1);
        const int pos1 = atomicAdd(&cnt[e1], 1);
        list[e0 * NTOK + pos0] = 2 * t;
        list[e1 * NTOK + pos1] = 2 * t + 1;
        prob[2 * t]     = p0;
        prob[2 * t + 1] = p1;
    }
}

// ---------------------------------------------------------------------------
// x (fp32) -> bf16, 4 elements/thread
// ---------------------------------------------------------------------------
__global__ __launch_bounds__(256) void cvt_x_kernel(
    const float* __restrict__ x, unsigned short* __restrict__ xb)
{
    const size_t i = ((size_t)blockIdx.x * 256 + threadIdx.x) * 4;
    const float4 v = *(const float4*)(x + i);
    ushort4 o;
    o.x = f2bf(v.x); o.y = f2bf(v.y); o.z = f2bf(v.z); o.w = f2bf(v.w);
    *(ushort4*)(xb + i) = o;
}

// ---------------------------------------------------------------------------
// Batched transpose+convert: in [R][C] fp32 -> out [C][R] bf16 (per expert).
// Makes weight layout N-major so GEMM B-tiles are K-contiguous; bf16 copy of
// both weight tensors (134 MB total) fits the 192 MB L2 for all GEMM re-reads.
// ---------------------------------------------------------------------------
__global__ void cvt_transpose_kernel(
    const float* __restrict__ in, unsigned short* __restrict__ out, int R, int C)
{
    __shared__ float tile[32][33];
    const size_t mat = (size_t)blockIdx.z * (size_t)R * C;
    const float* src = in + mat;
    unsigned short* dst = out + mat;
    const int c0 = blockIdx.x * 32, r0 = blockIdx.y * 32;
    const int tx = threadIdx.x, ty = threadIdx.y;   // 32 x 8
#pragma unroll
    for (int i = 0; i < 32; i += 8)
        tile[ty + i][tx] = src[(size_t)(r0 + ty + i) * C + c0 + tx];
    __syncthreads();
#pragma unroll
    for (int i = 0; i < 32; i += 8)
        dst[(size_t)(c0 + ty + i) * R + r0 + tx] = f2bf(tile[tx][ty + i]);
}

// ---------------------------------------------------------------------------
// Gather-GEMM with bf16 WMMA. Block tile 128(M) x 256(N), 8 waves as 2x4,
// wave tile 64x64 = 4x4 v_wmma_f32_16x16x32_bf16 per 32-K step (16 WMMAs per
// 16 ds_load_b128 -> best LDS-bandwidth-per-FLOP ratio). K-step 32.
// STAGE 1: h = gelu(Xg @ W1(N-major) + b1)  -> hb[pairslot][F] bf16
// STAGE 2: out[token] += p * (Hg @ W2(N-major) + b2)  (atomic f32, 2 adds/elt)
// ---------------------------------------------------------------------------
template<int STAGE, int KDIM>
__global__ __launch_bounds__(256) void moe_gemm_kernel(
    const unsigned short* __restrict__ Asrc,   // stage1: xb [NTOK][H]; stage2: hb [NPAIR][F]
    const unsigned short* __restrict__ Bsrc,   // stage1: w1b [E][F][H]; stage2: w2b [E][H][F]
    const float* __restrict__ bias,            // stage1: b1 [E][F]; stage2: b2 [E][H]
    const int* __restrict__ cnt,
    const int* __restrict__ list,
    const float* __restrict__ prob,
    unsigned short* __restrict__ hb,
    float* __restrict__ outp)
{
    constexpr int NDIM = (STAGE == 1) ? FDIM : HDIM;
    constexpr int LDR  = 40;                    // LDS row stride (bf16): 80B, conflict-free
    const int e  = blockIdx.z;
    const int m0 = blockIdx.x * 128;
    const int n0 = blockIdx.y * 256;
    const int count = cnt[e];
    if (m0 >= count) return;

    __shared__ __align__(16) unsigned short As[128 * LDR];   // 10.0 KB
    __shared__ __align__(16) unsigned short Bs[256 * LDR];   // 20.0 KB

    const int tid  = threadIdx.x;
    const int lane = tid & 31;
    const int wid  = tid >> 5;
    const int waveM = wid & 1;    // 2 waves along M (64 rows each)
    const int waveN = wid >> 1;   // 4 waves along N (64 cols each)

    // -------- global->LDS staging ------------------------------------------
    // A: 128 rows x 64B -> 2 threads/row, 32B each
    const int aRowId = tid >> 1;
    const int aHalf  = (tid & 1) * 16;
    const int aEntry = list[e * NTOK + min(m0 + aRowId, count - 1)];
    const size_t aRow  = (STAGE == 1) ? (size_t)(aEntry >> 1) : (size_t)aEntry;
    const size_t aBase = aRow * KDIM + aHalf;
    uint4* asDst = (uint4*)&As[aRowId * LDR + aHalf];
    // B: 256 rows x 64B -> 1 thread/row, 64B each
    const size_t bBase = (size_t)e * ((size_t)HDIM * FDIM)
                       + (size_t)(n0 + tid) * KDIM;
    uint4* bsDst = (uint4*)&Bs[tid * LDR];

    v8f acc[4][4];
#pragma unroll
    for (int i = 0; i < 4; ++i)
#pragma unroll
        for (int j = 0; j < 4; ++j) acc[i][j] = (v8f){0.f,0.f,0.f,0.f,0.f,0.f,0.f,0.f};

    const int fr = lane & 15;
    const int kb = (lane >> 4) * 8;             // per-lane K sub-offset (ISA 16-bit layout)

    for (int k0 = 0; k0 < KDIM; k0 += 32) {
        const uint4* ag = (const uint4*)(Asrc + aBase + k0);
        const uint4* bg = (const uint4*)(Bsrc + bBase + k0);
        const uint4 a0 = ag[0], a1 = ag[1];
        const uint4 b0 = bg[0], b1 = bg[1], b2 = bg[2], b3 = bg[3];
        __syncthreads();
        asDst[0] = a0; asDst[1] = a1;
        bsDst[0] = b0; bsDst[1] = b1; bsDst[2] = b2; bsDst[3] = b3;
        __syncthreads();

        FragU af[4], bf[4];
#pragma unroll
        for (int i = 0; i < 4; ++i) {
            const unsigned short* p = &As[(waveM * 64 + i * 16 + fr) * LDR + kb];
            af[i].u[0] = *(const uint4*)p;
            af[i].u[1] = *(const uint4*)(p + 16);
        }
#pragma unroll
        for (int j = 0; j < 4; ++j) {
            const unsigned short* p = &Bs[(waveN * 64 + j * 16 + fr) * LDR + kb];
            bf[j].u[0] = *(const uint4*)p;
            bf[j].u[1] = *(const uint4*)(p + 16);
        }
#pragma unroll
        for (int i = 0; i < 4; ++i)
#pragma unroll
            for (int j = 0; j < 4; ++j)
                acc[i][j] = __builtin_amdgcn_wmma_f32_16x16x32_bf16(
                    false, af[i].v, false, bf[j].v, (short)0, acc[i][j], false, false);
    }

    // -------- epilogue: C layout — vgpr v holds (m = v + 8*(lane>>4), n = lane&15)
    const int colBase = n0 + waveN * 64 + (lane & 15);
    const int mGroup  = (lane >> 4) * 8;
#pragma unroll
    for (int i = 0; i < 4; ++i) {
#pragma unroll
        for (int v = 0; v < 8; ++v) {
            const int rpos = m0 + waveM * 64 + i * 16 + mGroup + v;
            if (rpos >= count) continue;
            const int entry = list[e * NTOK + rpos];
#pragma unroll
            for (int j = 0; j < 4; ++j) {
                const int col = colBase + j * 16;
                const float val = acc[i][j][v] + bias[e * NDIM + col];
                if (STAGE == 1) {
                    const float g = 0.5f * val * (1.f + erff(val * 0.7071067811865475f));
                    hb[(size_t)entry * FDIM + col] = f2bf(g);
                } else {
                    const float p = prob[entry];
                    atomicAdd(&outp[(size_t)(entry >> 1) * HDIM + col], p * val);
                }
            }
        }
    }
}

// ---------------------------------------------------------------------------
extern "C" void kernel_launch(void* const* d_in, const int* in_sizes, int n_in,
                              void* d_out, int out_size, void* d_ws, size_t ws_size,
                              hipStream_t stream)
{
    (void)in_sizes; (void)n_in; (void)ws_size;
    const float* x        = (const float*)d_in[0];
    const float* router_w = (const float*)d_in[1];
    const float* router_b = (const float*)d_in[2];
    const float* w1       = (const float*)d_in[3];
    const float* b1       = (const float*)d_in[4];
    const float* w2       = (const float*)d_in[5];
    const float* b2       = (const float*)d_in[6];
    // d_in[7] = k (==2, hardcoded)

    char* ws = (char*)d_ws;
    size_t off = 0;
    int*            cnt  = (int*)(ws + off);            off += 256;
    int*            list = (int*)(ws + off);            off += (size_t)NEXP * NTOK * 4;
    float*          prob = (float*)(ws + off);          off += (size_t)NPAIR * 4;
    unsigned short* xb   = (unsigned short*)(ws + off); off += (size_t)NTOK * HDIM * 2;
    unsigned short* w1b  = (unsigned short*)(ws + off); off += (size_t)NEXP * HDIM * FDIM * 2;
    unsigned short* w2b  = (unsigned short*)(ws + off); off += (size_t)NEXP * HDIM * FDIM * 2;
    unsigned short* hb   = (unsigned short*)(ws + off); off += (size_t)NPAIR * FDIM * 2;

    hipMemsetAsync(cnt, 0, 256, stream);
    hipMemsetAsync(d_out, 0, (size_t)out_size * sizeof(float), stream);

    router_kernel<<<NTOK / 8, 256, 0, stream>>>(x, router_w, router_b, cnt, list, prob);
    cvt_x_kernel<<<((size_t)NTOK * HDIM / 4 + 255) / 256, 256, 0, stream>>>(x, xb);

    dim3 tb(32, 8, 1);
    cvt_transpose_kernel<<<dim3(FDIM / 32, HDIM / 32, NEXP), tb, 0, stream>>>(w1, w1b, HDIM, FDIM);
    cvt_transpose_kernel<<<dim3(HDIM / 32, FDIM / 32, NEXP), tb, 0, stream>>>(w2, w2b, FDIM, HDIM);

    // Stage 1: [count_e,1024] x [1024,4096] per expert, GELU epilogue
    moe_gemm_kernel<1, HDIM><<<dim3(NTOK / 128, FDIM / 256, NEXP), 256, 0, stream>>>(
        xb, w1b, b1, cnt, list, prob, hb, nullptr);
    // Stage 2: [count_e,4096] x [4096,1024] per expert, prob-weighted scatter-add
    moe_gemm_kernel<2, FDIM><<<dim3(NTOK / 128, HDIM / 256, NEXP), 256, 0, stream>>>(
        hb, w2b, b2, cnt, list, prob, nullptr, (float*)d_out);
}
